// GRU_Serial_34832184771262
// MI455X (gfx1250) — matrix operand order repaired and checked by reading, not compiled, not run on hardware
//
#include <hip/hip_runtime.h>

// ---------------------------------------------------------------------------
// 2-layer GRU, persistent-wave pipelined implementation for gfx1250 (MI455X).
//   B=64, T=4096, D=H=128.  grid=2 blocks (block 0 = layer 0, block 1 = layer 1)
//   block = 256 threads = 8 waves (wave32). Wave w owns hidden columns
//   [16w, 16w+16). WMMA: v_wmma_f32_16x16x32_bf16, f32 accumulate.
// ---------------------------------------------------------------------------

typedef __attribute__((ext_vector_type(16))) __bf16        v16bf;
typedef __attribute__((ext_vector_type(8)))  float         v8f;
typedef __attribute__((ext_vector_type(4)))  unsigned int  u32x4;

static constexpr int Tn   = 4096;
static constexpr int Bn   = 64;
static constexpr int Dn   = 128;
static constexpr int Hn   = 128;
static constexpr int RING = 256;                // ring slots for h0 stream
static constexpr int WELEM = 24 * 8 * 32 * 16;  // packed fused weight elements (bf16)

union Frag { v16bf v; u32x4 u[2]; };

__device__ __forceinline__ unsigned short f2bf(float f) {
  unsigned int u = __float_as_uint(f);
  u += 0x7FFFu + ((u >> 16) & 1u);              // round-to-nearest-even
  return (unsigned short)(u >> 16);
}

__device__ __forceinline__ float sigm(float x) {
  return 1.0f / (1.0f + __expf(-x));
}

// A operand (16x32 bf16): row-major [64 x 128] bf16 source, row stride 128.
// Per ISA layout: lane = M%16 (+16 for K-half), elems 0..7 -> K=lo..lo+7,
// elems 8..15 -> K=16+lo..16+lo+7, lo = (lane>=16)?8:0.  Two 16B loads.
__device__ __forceinline__ v16bf load_afrag(const unsigned short* base, int mt,
                                            int kcol, int lane) {
  const int row = mt * 16 + (lane & 15);
  const int lo  = (lane >> 4) << 3;
  const unsigned short* p = base + row * Hn + kcol + lo;
  Frag f;
  f.u[0] = *reinterpret_cast<const u32x4*>(p);
  f.u[1] = *reinterpret_cast<const u32x4*>(p + 16);
  return f.v;
}

// B operand from pre-packed LDS: fragment (nt, kc) is 32 lanes x 32 bytes.
__device__ __forceinline__ v16bf load_bfrag(const unsigned short* wp, int nt,
                                            int kc, int lane) {
  const unsigned short* p = wp + (((nt * 8 + kc) * 32) + lane) * 16;
  Frag f;
  f.u[0] = *reinterpret_cast<const u32x4*>(p);
  f.u[1] = *reinterpret_cast<const u32x4*>(p + 8);
  return f.v;
}

extern "C" __global__ __launch_bounds__(256, 1)
void gru_pipeline_kernel(const float* __restrict__ x,
                         const float* __restrict__ Wih0, const float* __restrict__ Whh0,
                         const float* __restrict__ bih0, const float* __restrict__ bhh0,
                         const float* __restrict__ Wih1, const float* __restrict__ Whh1,
                         const float* __restrict__ bih1, const float* __restrict__ bhh1,
                         float* __restrict__ out,
                         unsigned short* __restrict__ ring,   // [RING][Bn*Hn] bf16
                         int* __restrict__ prod,              // [RING]
                         int* __restrict__ cons)              // [RING]
{
  // 192KB packed weights + 16KB input stage + 16KB h stage = 224KB LDS / WG.
  __shared__ unsigned short wpack[WELEM];     // fused [Wih|Whh]^T, B-frag layout
  __shared__ unsigned short a_in[Bn * Dn];    // x_t (layer0) / h0_t (layer1), bf16
  __shared__ unsigned short h_buf[Bn * Hn];   // own h_{t-1}, bf16

  const int layer = blockIdx.x;
  const int tid   = threadIdx.x;
  const int lane  = tid & 31;
  const int wave  = tid >> 5;

  const float* Wih = layer ? Wih1 : Wih0;
  const float* Whh = layer ? Whh1 : Whh0;
  const float* bih = layer ? bih1 : bih0;
  const float* bhh = layer ? bhh1 : bhh0;

  // ---- one-time: pack fused weight [3H x (D+H)] into WMMA-B layout (bf16) ---
  for (int idx = tid; idx < WELEM; idx += 256) {
    const int e  = idx & 15;
    const int ll = (idx >> 4) & 31;
    const int kc = (idx >> 9) & 7;
    const int nt = idx >> 12;
    const int n  = nt * 16 + (ll & 15);
    const int K  = kc * 32 + ((ll >> 4) << 3) + ((e >> 3) << 4) + (e & 7);   // lanes 16-31 hold K+8
    const float v = (K < 128) ? Wih[n * 128 + K] : Whh[n * 128 + (K - 128)];
    wpack[idx] = f2bf(v);
  }
  for (int idx = tid; idx < Bn * Hn; idx += 256) h_buf[idx] = 0;  // h_0 = 0

  // per-lane gate biases for owned column j
  const int j = wave * 16 + (lane & 15);
  const float bR  = bih[j]       + bhh[j];
  const float bZ  = bih[128 + j] + bhh[128 + j];
  const float bIN = bih[256 + j];
  const float bHN = bhh[256 + j];

  // persistent f32 hidden state (this wave owns exactly these elements)
  v8f hreg[4];
#pragma unroll
  for (int mt = 0; mt < 4; ++mt)
#pragma unroll
    for (int i = 0; i < 8; ++i) hreg[mt][i] = 0.0f;

  __syncthreads();

  for (int t = 0; t < Tn; ++t) {
    const int slot = t & (RING - 1);

    // ------------------ stage this step's input into a_in ------------------
    if (layer == 0) {
#pragma unroll 4
      for (int e = 0; e < 32; ++e) {
        const int idx = tid + e * 256;             // 0..8191
        const int b = idx >> 7, d = idx & 127;
        a_in[idx] = f2bf(x[((size_t)b * Tn + t) * Dn + d]);
      }
      if (t + 1 < Tn)                              // warm L2 for next step
        __builtin_prefetch(&x[((size_t)(tid >> 2) * Tn + (t + 1)) * Dn], 0, 0);
      if (t >= RING && tid == 0) {                 // ring backpressure
        while (__hip_atomic_load(cons + slot, __ATOMIC_ACQUIRE,
                                 __HIP_MEMORY_SCOPE_AGENT) < t - RING + 1)
          __builtin_amdgcn_s_sleep(1);
      }
    } else {
      if (tid == 0) {                              // wait for h0[t]
        while (__hip_atomic_load(prod + slot, __ATOMIC_ACQUIRE,
                                 __HIP_MEMORY_SCOPE_AGENT) != t + 1)
          __builtin_amdgcn_s_sleep(1);
      }
      __syncthreads();
      const u32x4* src = reinterpret_cast<const u32x4*>(ring + (size_t)slot * Bn * Hn);
      u32x4*       dst = reinterpret_cast<u32x4*>(a_in);
#pragma unroll
      for (int e = 0; e < 4; ++e) dst[tid + e * 256] = src[tid + e * 256];
    }
    __syncthreads();
    if (layer == 1 && tid == 0)                    // slot consumed
      __hip_atomic_store(cons + slot, t + 1, __ATOMIC_RELEASE,
                         __HIP_MEMORY_SCOPE_AGENT);

    // ------------------ fused gate GEMMs (K = 256 = [x_t | h]) -------------
    v8f aR[4], aZ[4], aIN[4], aHN[4];
#pragma unroll
    for (int mt = 0; mt < 4; ++mt)
#pragma unroll
      for (int i = 0; i < 8; ++i) { aR[mt][i] = 0.f; aZ[mt][i] = 0.f;
                                    aIN[mt][i] = 0.f; aHN[mt][i] = 0.f; }

#pragma unroll
    for (int kc = 0; kc < 8; ++kc) {
      const v16bf bfR = load_bfrag(wpack,      wave, kc, lane);   // r columns
      const v16bf bfZ = load_bfrag(wpack,  8 + wave, kc, lane);   // z columns
      const v16bf bfN = load_bfrag(wpack, 16 + wave, kc, lane);   // n columns
      const unsigned short* asrc = (kc < 4) ? a_in : h_buf;
      const int kcol = (kc & 3) * 32;
#pragma unroll
      for (int mt = 0; mt < 4; ++mt) {
        const v16bf af = load_afrag(asrc, mt, kcol, lane);
        aR[mt] = __builtin_amdgcn_wmma_f32_16x16x32_bf16(
            false, af, false, bfR, (short)0, aR[mt], false, false);
        aZ[mt] = __builtin_amdgcn_wmma_f32_16x16x32_bf16(
            false, af, false, bfZ, (short)0, aZ[mt], false, false);
        if (kc < 4)
          aIN[mt] = __builtin_amdgcn_wmma_f32_16x16x32_bf16(
              false, af, false, bfN, (short)0, aIN[mt], false, false);
        else
          aHN[mt] = __builtin_amdgcn_wmma_f32_16x16x32_bf16(
              false, af, false, bfN, (short)0, aHN[mt], false, false);
      }
    }
    __syncthreads();   // everyone done reading h_buf before it is rewritten

    // ------------------ gates + h update (wave-local) ----------------------
#pragma unroll
    for (int mt = 0; mt < 4; ++mt) {
#pragma unroll
      for (int i = 0; i < 8; ++i) {
        const float r = sigm(aR[mt][i] + bR);
        const float z = sigm(aZ[mt][i] + bZ);
        const float n = tanhf(aIN[mt][i] + bIN + r * (aHN[mt][i] + bHN));
        const float h = (1.0f - z) * n + z * hreg[mt][i];
        hreg[mt][i] = h;
        const int M = mt * 16 + i + ((lane >> 4) << 3);   // C/D layout row
        const unsigned short hb = f2bf(h);
        h_buf[M * Hn + j] = hb;
        if (layer == 0)
          ring[(size_t)slot * Bn * Hn + M * Hn + j] = hb;
      }
    }
    if (layer == 0) __threadfence();               // ring writes visible
    __syncthreads();
    if (layer == 0 && tid == 0)
      __hip_atomic_store(prod + slot, t + 1, __ATOMIC_RELEASE,
                         __HIP_MEMORY_SCOPE_AGENT);
  }

  // ------------------ final hidden state -> out[layer, B, H] ---------------
#pragma unroll
  for (int mt = 0; mt < 4; ++mt)
#pragma unroll
    for (int i = 0; i < 8; ++i) {
      const int M = mt * 16 + i + ((lane >> 4) << 3);
      out[layer * Bn * Hn + M * Hn + j] = hreg[mt][i];
    }
}

// ---------------------------------------------------------------------------
extern "C" void kernel_launch(void* const* d_in, const int* in_sizes, int n_in,
                              void* d_out, int out_size, void* d_ws, size_t ws_size,
                              hipStream_t stream) {
  (void)in_sizes; (void)n_in; (void)out_size; (void)ws_size;
  const float* x    = (const float*)d_in[0];
  const float* Wih0 = (const float*)d_in[1];
  const float* Whh0 = (const float*)d_in[2];
  const float* bih0 = (const float*)d_in[3];
  const float* bhh0 = (const float*)d_in[4];
  const float* Wih1 = (const float*)d_in[5];
  const float* Whh1 = (const float*)d_in[6];
  const float* bih1 = (const float*)d_in[7];
  const float* bhh1 = (const float*)d_in[8];

  // ws layout: [ring bf16: RING*Bn*Hn ushorts = 4MB][prod: RING ints][cons: RING ints]
  unsigned short* ring = (unsigned short*)d_ws;
  int* prod = (int*)((char*)d_ws + (size_t)RING * Bn * Hn * sizeof(unsigned short));
  int* cons = prod + RING;

  // Reset flags every call (graph-capture-legal) so the produce/consume
  // protocol is deterministic across replays.
  hipMemsetAsync(prod, 0, 2 * RING * sizeof(int), stream);

  hipLaunchKernelGGL(gru_pipeline_kernel, dim3(2), dim3(256), 0, stream,
                     x, Wih0, Whh0, bih0, bhh0, Wih1, Whh1, bih1, bhh1,
                     (float*)d_out, ring, prod, cons);
}
